// MultiheadAttention_2791728742724
// MI455X (gfx1250) — matrix operand order repaired and checked by reading, not compiled, not run on hardware
//
#include <hip/hip_runtime.h>
#include <hip/hip_bf16.h>

// ---------------------------------------------------------------------------
// MI455X (gfx1250, wave32) fused multi-head attention forward.
// All matmuls via v_wmma_f32_16x16x32_bf16, flash-style streaming softmax.
// ---------------------------------------------------------------------------

typedef __bf16 bf16;
typedef __attribute__((ext_vector_type(16))) __bf16 v16bf;
typedef __attribute__((ext_vector_type(8)))  __bf16 bf16x8;
typedef __attribute__((ext_vector_type(4)))  __bf16 bf16x4;
typedef __attribute__((ext_vector_type(8)))  float  v8f;

#define DEVINL __device__ __forceinline__

constexpr int Bb    = 2;
constexpr int S     = 2048;
constexpr int E     = 1024;
constexpr int H     = 16;
constexpr int Dh    = 64;          // head dim
constexpr int M_TOT = Bb * S;      // 4096 rows
constexpr int N_QKV = 3 * E;       // 3072

DEVINL int lane_id() { return threadIdx.x & 31; }

DEVINL v8f v8f_zero() {
  v8f z;
#pragma unroll
  for (int e = 0; e < 8; ++e) z[e] = 0.0f;
  return z;
}

// A-fragment (16x32 bf16, M x K), §7.12.2: lanes 0-15 row M=lane, K {0..7,16..23};
// lanes 16-31 same rows, K {8..15,24..31}. Source row-major [*, ld].
DEVINL v16bf load_a_frag(const bf16* __restrict__ base, int ld, int row, int k) {
  const int lane = lane_id();
  const int r    = row + (lane & 15);
  const int kk   = k + ((lane >> 4) << 3);      // +8 for upper lane half
  union { v16bf v; bf16x8 h[2]; } u;
  u.h[0] = *(const bf16x8*)(base + (size_t)r * ld + kk);
  u.h[1] = *(const bf16x8*)(base + (size_t)r * ld + kk + 16);
  return u.v;
}

// B-fragment (32x16 bf16, K x N) where B[k][n] = src[n][k] (e.g. torch Linear
// weight [out,in], or V^T). Lane holds column n=lane&15, K {0..7,16..23}
// (lanes 16-31: K {8..15,24..31}) -> two contiguous 16B chunks per lane.
DEVINL v16bf load_bT_frag(const bf16* __restrict__ wt, int ld, int ncol, int k) {
  const int lane = lane_id();
  const int n    = ncol + (lane & 15);
  const int kk   = k + ((lane >> 4) << 3);      // +8 for upper lane half
  union { v16bf v; bf16x8 h[2]; } u;
  u.h[0] = *(const bf16x8*)(wt + (size_t)n * ld + kk);
  u.h[1] = *(const bf16x8*)(wt + (size_t)n * ld + kk + 16);
  return u.v;
}

DEVINL v8f wmma_bf16(v16bf a, v16bf b, v8f c) {
  return __builtin_amdgcn_wmma_f32_16x16x32_bf16(
      /*neg_a=*/false, a, /*neg_b=*/false, b,
      /*c_mod=*/(short)0, c, /*reuse_a=*/false, /*reuse_b=*/false);
}

// ---------------------------------------------------------------------------
// fp32 -> bf16 bulk convert (n4 = count/4)
// ---------------------------------------------------------------------------
__global__ void f32_to_bf16_kernel(const float* __restrict__ in,
                                   bf16* __restrict__ out, int n4) {
  int i = blockIdx.x * blockDim.x + threadIdx.x;
  if (i < n4) {
    float4 v = ((const float4*)in)[i];
    bf16x4 o;
    o[0] = (bf16)v.x; o[1] = (bf16)v.y; o[2] = (bf16)v.z; o[3] = (bf16)v.w;
    ((bf16x4*)out)[i] = o;
  }
}

// ---------------------------------------------------------------------------
// QKV projection: [4096,1024] x [3072,1024]^T -> scatter into Q,K (row-major
// per (b,h)) and V transposed (d-major) for contiguous PV B-fragments.
// One wave computes a 64x64 tile; block = 4 waves along N (64 x 256 tile).
// ---------------------------------------------------------------------------
__global__ __launch_bounds__(128) void qkv_gemm_kernel(
    const bf16* __restrict__ X, const bf16* __restrict__ W,
    const float* __restrict__ bias,
    bf16* __restrict__ Qb, bf16* __restrict__ Kb, bf16* __restrict__ Vt) {
  const int wid  = threadIdx.x >> 5;
  const int lane = lane_id();
  const int n0   = blockIdx.x * 256 + wid * 64;
  const int m0   = blockIdx.y * 64;

  v8f acc[4][4];
#pragma unroll
  for (int mi = 0; mi < 4; ++mi)
#pragma unroll
    for (int ni = 0; ni < 4; ++ni) acc[mi][ni] = v8f_zero();

  for (int k = 0; k < E; k += 32) {
    v16bf a[4], b[4];
#pragma unroll
    for (int mi = 0; mi < 4; ++mi) a[mi] = load_a_frag(X, E, m0 + mi * 16, k);
#pragma unroll
    for (int ni = 0; ni < 4; ++ni) b[ni] = load_bT_frag(W, E, n0 + ni * 16, k);
#pragma unroll
    for (int mi = 0; mi < 4; ++mi)
#pragma unroll
      for (int ni = 0; ni < 4; ++ni)
        acc[mi][ni] = wmma_bf16(a[mi], b[ni], acc[mi][ni]);
  }

  const int grp = lane >> 4;
  const int cl  = lane & 15;
#pragma unroll
  for (int mi = 0; mi < 4; ++mi)
#pragma unroll
    for (int ni = 0; ni < 4; ++ni)
#pragma unroll
      for (int i = 0; i < 8; ++i) {
        int   m = m0 + mi * 16 + i + grp * 8;
        int   n = n0 + ni * 16 + cl;
        float v = acc[mi][ni][i] + bias[n];
        bf16  hv = (bf16)v;
        int   bq = m >> 11;            // / S
        int   s  = m & (S - 1);
        int   h  = n / (3 * Dh);
        int   r  = n % (3 * Dh);
        int   bh = bq * H + h;
        if (r < Dh)
          Qb[((size_t)bh * S + s) * Dh + r] = hv;
        else if (r < 2 * Dh)
          Kb[((size_t)bh * S + s) * Dh + (r - Dh)] = hv;
        else
          Vt[((size_t)bh * Dh + (r - 2 * Dh)) * S + s] = hv;
      }
}

// ---------------------------------------------------------------------------
// Flash attention: one wave handles (b,h, 32-row query block); streams key
// blocks of 32 with running max/sum; P staged through per-wave LDS tile for
// C-layout f32 -> A-layout bf16 conversion feeding the PV WMMAs.
// ---------------------------------------------------------------------------
__global__ __launch_bounds__(128) void flash_attn_kernel(
    const bf16* __restrict__ Qb, const bf16* __restrict__ Kb,
    const bf16* __restrict__ Vt, bf16* __restrict__ Yv) {
  __shared__ bf16 plds[4][32 * 32];   // 2 KB per wave

  const int wid  = threadIdx.x >> 5;
  const int lane = lane_id();
  const int grp  = lane >> 4;
  const int cl   = lane & 15;
  const int wg   = blockIdx.x * 4 + wid;
  const int bh   = wg >> 6;           // 0..31
  const int qb   = wg & 63;           // 0..63
  const int q0   = qb * 32;
  const int bq   = bh >> 4;
  const int h    = bh & 15;

  const bf16* qp = Qb + (size_t)bh * S * Dh;
  const bf16* kp = Kb + (size_t)bh * S * Dh;
  const bf16* vp = Vt + (size_t)bh * Dh * S;
  bf16* pw = &plds[wid][0];

  // Q fragments held resident: 2 row-tiles x 2 K-chunks over d=64
  v16bf qf[2][2];
#pragma unroll
  for (int mi = 0; mi < 2; ++mi)
#pragma unroll
    for (int kc = 0; kc < 2; ++kc)
      qf[mi][kc] = load_a_frag(qp, Dh, q0 + mi * 16, kc * 32);

  v8f  oacc[2][4];
  float mrun[2][8], lrun[2][8];
#pragma unroll
  for (int mi = 0; mi < 2; ++mi) {
#pragma unroll
    for (int ni = 0; ni < 4; ++ni) oacc[mi][ni] = v8f_zero();
#pragma unroll
    for (int i = 0; i < 8; ++i) { mrun[mi][i] = -INFINITY; lrun[mi][i] = 0.0f; }
  }

  const float scale = 0.125f;   // 1/sqrt(64)

  for (int kb = 0; kb < S; kb += 32) {
    // ---- logits tile S32x32 = Q(32x64) @ K^T(64x32) ----
    v8f sc[2][2];
#pragma unroll
    for (int mi = 0; mi < 2; ++mi)
#pragma unroll
      for (int ni = 0; ni < 2; ++ni) sc[mi][ni] = v8f_zero();

#pragma unroll
    for (int kc = 0; kc < 2; ++kc) {
      v16bf kf[2];
#pragma unroll
      for (int ni = 0; ni < 2; ++ni)
        kf[ni] = load_bT_frag(kp, Dh, kb + ni * 16, kc * 32);
#pragma unroll
      for (int mi = 0; mi < 2; ++mi)
#pragma unroll
        for (int ni = 0; ni < 2; ++ni)
          sc[mi][ni] = wmma_bf16(qf[mi][kc], kf[ni], sc[mi][ni]);
    }

    // ---- streaming softmax update (per C-layout row group) ----
#pragma unroll
    for (int mi = 0; mi < 2; ++mi) {
      float alpha[8];
#pragma unroll
      for (int i = 0; i < 8; ++i) {
        float t0 = sc[mi][0][i] * scale;
        float t1 = sc[mi][1][i] * scale;
        float rm = fmaxf(t0, t1);
        rm = fmaxf(rm, __shfl_xor(rm, 1, 32));
        rm = fmaxf(rm, __shfl_xor(rm, 2, 32));
        rm = fmaxf(rm, __shfl_xor(rm, 4, 32));
        rm = fmaxf(rm, __shfl_xor(rm, 8, 32));       // row max within 16-lane group
        float mnew = fmaxf(mrun[mi][i], rm);
        float a    = __expf(mrun[mi][i] - mnew);
        mrun[mi][i] = mnew;
        float p0 = __expf(t0 - mnew);
        float p1 = __expf(t1 - mnew);
        sc[mi][0][i] = p0;
        sc[mi][1][i] = p1;
        float rs = p0 + p1;
        rs += __shfl_xor(rs, 1, 32);
        rs += __shfl_xor(rs, 2, 32);
        rs += __shfl_xor(rs, 4, 32);
        rs += __shfl_xor(rs, 8, 32);
        lrun[mi][i] = lrun[mi][i] * a + rs;
        alpha[i]    = a;
      }
#pragma unroll
      for (int ni = 0; ni < 4; ++ni)
#pragma unroll
        for (int i = 0; i < 8; ++i) oacc[mi][ni][i] *= alpha[i];

      // P (C layout) -> LDS row-major bf16 [qrow][key]
#pragma unroll
      for (int ni = 0; ni < 2; ++ni)
#pragma unroll
        for (int i = 0; i < 8; ++i)
          pw[(mi * 16 + i + grp * 8) * 32 + ni * 16 + cl] = (bf16)sc[mi][ni][i];
    }

    asm volatile("s_wait_dscnt 0" ::: "memory");   // LDS stores visible to ds_loads

    // ---- O += P(32x32) @ V(32x64) ----
    v16bf af[2];
#pragma unroll
    for (int mi = 0; mi < 2; ++mi) af[mi] = load_a_frag(pw, 32, mi * 16, 0);
#pragma unroll
    for (int ni = 0; ni < 4; ++ni) {
      v16bf vf = load_bT_frag(vp, S, ni * 16, kb);  // V^T: contiguous keys per lane
#pragma unroll
      for (int mi = 0; mi < 2; ++mi)
        oacc[mi][ni] = wmma_bf16(af[mi], vf, oacc[mi][ni]);
    }
    asm volatile("" ::: "memory");  // keep P loads ordered before next iter's stores
  }

  // ---- epilogue: normalize, write [B,S,E] bf16 ----
#pragma unroll
  for (int mi = 0; mi < 2; ++mi)
#pragma unroll
    for (int ni = 0; ni < 4; ++ni)
#pragma unroll
      for (int i = 0; i < 8; ++i) {
        int   srow = q0 + mi * 16 + i + grp * 8;
        int   dcol = ni * 16 + cl;
        float v    = oacc[mi][ni][i] / lrun[mi][i];
        Yv[((size_t)(bq * S + srow)) * E + h * Dh + dcol] = (bf16)v;
      }
}

// ---------------------------------------------------------------------------
// Output projection: [4096,1024] x [1024,1024]^T + bias -> fp32 out
// ---------------------------------------------------------------------------
__global__ __launch_bounds__(128) void out_gemm_kernel(
    const bf16* __restrict__ A, const bf16* __restrict__ W,
    const float* __restrict__ bias, float* __restrict__ Out) {
  const int wid  = threadIdx.x >> 5;
  const int lane = lane_id();
  const int n0   = blockIdx.x * 256 + wid * 64;
  const int m0   = blockIdx.y * 64;

  v8f acc[4][4];
#pragma unroll
  for (int mi = 0; mi < 4; ++mi)
#pragma unroll
    for (int ni = 0; ni < 4; ++ni) acc[mi][ni] = v8f_zero();

  for (int k = 0; k < E; k += 32) {
    v16bf a[4], b[4];
#pragma unroll
    for (int mi = 0; mi < 4; ++mi) a[mi] = load_a_frag(A, E, m0 + mi * 16, k);
#pragma unroll
    for (int ni = 0; ni < 4; ++ni) b[ni] = load_bT_frag(W, E, n0 + ni * 16, k);
#pragma unroll
    for (int mi = 0; mi < 4; ++mi)
#pragma unroll
      for (int ni = 0; ni < 4; ++ni)
        acc[mi][ni] = wmma_bf16(a[mi], b[ni], acc[mi][ni]);
  }

  const int grp = lane >> 4;
  const int cl  = lane & 15;
#pragma unroll
  for (int mi = 0; mi < 4; ++mi)
#pragma unroll
    for (int ni = 0; ni < 4; ++ni)
#pragma unroll
      for (int i = 0; i < 8; ++i) {
        int m = m0 + mi * 16 + i + grp * 8;
        int n = n0 + ni * 16 + cl;
        Out[(size_t)m * E + n] = acc[mi][ni][i] + bias[n];
      }
}

// ---------------------------------------------------------------------------
// Launch
// ---------------------------------------------------------------------------
extern "C" void kernel_launch(void* const* d_in, const int* in_sizes, int n_in,
                              void* d_out, int out_size, void* d_ws, size_t ws_size,
                              hipStream_t stream) {
  const float* x_f  = (const float*)d_in[0];   // [2,2048,1024]
  const float* qw_f = (const float*)d_in[1];   // [3072,1024]
  const float* qb_f = (const float*)d_in[2];   // [3072]
  const float* ow_f = (const float*)d_in[3];   // [1024,1024]
  const float* ob_f = (const float*)d_in[4];   // [1024]

  char*  ws  = (char*)d_ws;
  size_t off = 0;
  auto alloc = [&](size_t bytes) -> void* {
    void* p = ws + off;
    off += (bytes + 255) & ~(size_t)255;
    return p;
  };
  bf16* xb  = (bf16*)alloc((size_t)M_TOT * E * 2);       // 8 MB
  bf16* qwb = (bf16*)alloc((size_t)N_QKV * E * 2);       // 6 MB
  bf16* owb = (bf16*)alloc((size_t)E * E * 2);           // 2 MB
  bf16* Qb  = (bf16*)alloc((size_t)Bb * H * S * Dh * 2); // 8 MB
  bf16* Kb  = (bf16*)alloc((size_t)Bb * H * S * Dh * 2); // 8 MB
  bf16* Vt  = (bf16*)alloc((size_t)Bb * H * S * Dh * 2); // 8 MB
  bf16* Yv  = (bf16*)alloc((size_t)M_TOT * E * 2);       // 8 MB
  (void)ws_size; (void)in_sizes; (void)n_in; (void)out_size;

  int n4;
  n4 = M_TOT * E / 4;
  f32_to_bf16_kernel<<<(n4 + 255) / 256, 256, 0, stream>>>(x_f, xb, n4);
  n4 = N_QKV * E / 4;
  f32_to_bf16_kernel<<<(n4 + 255) / 256, 256, 0, stream>>>(qw_f, qwb, n4);
  n4 = E * E / 4;
  f32_to_bf16_kernel<<<(n4 + 255) / 256, 256, 0, stream>>>(ow_f, owb, n4);

  qkv_gemm_kernel<<<dim3(N_QKV / 256, M_TOT / 64), 128, 0, stream>>>(
      xb, qwb, qb_f, Qb, Kb, Vt);

  flash_attn_kernel<<<(Bb * H * (S / 32)) / 4, 128, 0, stream>>>(Qb, Kb, Vt, Yv);

  out_gemm_kernel<<<dim3(E / 256, M_TOT / 64), 128, 0, stream>>>(
      Yv, owb, ob_f, (float*)d_out);
}
